// GNN_basis_11003706213268
// MI455X (gfx1250) — matrix operand order repaired and checked by reading, not compiled, not run on hardware
//
#include <hip/hip_runtime.h>
#include <cstdint>
#include <cstddef>

#ifndef USE_ASYNC
#define USE_ASYNC 1
#endif

typedef unsigned short u16;
typedef __attribute__((ext_vector_type(16))) __bf16 v16bf;
typedef __attribute__((ext_vector_type(8)))  float  v8f;

#define F_  384
#define V_  128

static __device__ __forceinline__ u16 f32_to_bf16(float f) {
    unsigned int u = __float_as_uint(f);
    u += 0x7FFFu + ((u >> 16) & 1u);          // round-to-nearest-even
    return (u16)(u >> 16);
}

static __device__ __forceinline__ float swishf(float v) {
    return v / (1.0f + __expf(-v));
}

// Wave-relative LDS byte offset: generic LDS-aperture pointers carry the
// LDS offset in their low 32 bits (ISA 10.2).
static __device__ __forceinline__ unsigned lds_off(const void* p) {
    return (unsigned)(unsigned long long)(uintptr_t)p;
}

#if USE_ASYNC
// GLOBAL_LOAD_ASYNC_TO_LDS_B128: per-lane 16B global -> LDS[ldsOff], ASYNCcnt.
static __device__ __forceinline__ void async_b128(unsigned ldsOff, const void* gp) {
    unsigned long long ga = (unsigned long long)(uintptr_t)gp;
    asm volatile("global_load_async_to_lds_b128 %0, %1, off"
                 :: "v"(ldsOff), "v"(ga) : "memory");
}
static __device__ __forceinline__ void wait_async0() {
    asm volatile("s_wait_asynccnt 0x0" ::: "memory");
}
#endif

// ------------------------------------------------------------------
// WMMA GEMM:  C(MxN) = epilogue(A(MxK,bf16) * B(KxN) + bias)
// B pre-transposed as Bt (N x K bf16) -> all global/LDS traffic is b128.
// Block: 256 threads (8 waves) -> 128(M) x 128(N) tile, K-step 32.
// Waves: 4 in M (32 rows each: 2 A-frags) x 2 in N (64 cols: 4 B-frags)
//        => 8 v_wmma per wave per K-step.
// Double-buffered LDS, tiles filled with async loads (ASYNCcnt).
// All 12 fragment ds_loads issued before the 8 WMMAs so the compiler can
// use partial s_wait_dscnt and keep the matrix pipe busy.
// A tail rows are CLAMPED (not predicated) -> branch-free hot loop;
// epilogue masks out-of-range rows.
// MODE: 0 swish->bf16, 1 swish->f32, 2 linear->f32, 3 f32 C += swish(v)
// ------------------------------------------------------------------
#define BM 128
#define BNC 128
#define BK 32
#define LDT 40   // 32 + 8 pad: 80B row stride = 20-bank stride, conflict-free

template<int MODE>
__global__ __launch_bounds__(256)
void wmma_gemm(const u16* __restrict__ A, const u16* __restrict__ Bt,
               const float* __restrict__ bias, void* __restrict__ Cout,
               int M, int N, int K)
{
    __shared__ alignas(16) u16 sA[2][BM * LDT];
    __shared__ alignas(16) u16 sB[2][BNC * LDT];

    const int tid   = threadIdx.x;
    const int wave  = tid >> 5;
    const int lane  = tid & 31;
    const int waveM = wave & 3;        // 4 M-groups of 32 rows
    const int waveN = wave >> 2;       // 2 N-groups of 64 cols
    const int m0 = blockIdx.x * BM;
    const int n0 = blockIdx.y * BNC;

    v8f acc[2][4] = {};

    const int hi = lane >> 4;
    const int mA = lane & 15;
    const int nB = lane & 15;
    const int kA = hi * 8;             // A frag: chunks at +kA, +kA+16
    const int kB = hi * 8;             // B frag: chunks at +kB, +kB+16

    // ---- tile-fill mapping: 512 uint4 per tile, 2 per thread ----
    const int uA0 = tid, uA1 = tid + 256;
    const int rA0 = uA0 >> 2, cA0 = uA0 & 3;
    const int rA1 = uA1 >> 2, cA1 = uA1 & 3;
    int grA0 = m0 + rA0; if (grA0 > M - 1) grA0 = M - 1;   // clamp tail
    int grA1 = m0 + rA1; if (grA1 > M - 1) grA1 = M - 1;
    const u16* aP0 = A + (size_t)grA0 * K + cA0 * 8;
    const u16* aP1 = A + (size_t)grA1 * K + cA1 * 8;
    const u16* bP0 = Bt + (size_t)(n0 + rA0) * K + cA0 * 8;
    const u16* bP1 = Bt + (size_t)(n0 + rA1) * K + cA1 * 8;
    const unsigned stA0 = (unsigned)(rA0 * LDT + cA0 * 8) * 2u;
    const unsigned stA1 = (unsigned)(rA1 * LDT + cA1 * 8) * 2u;
    const unsigned sAb[2] = { lds_off(&sA[0][0]), lds_off(&sA[1][0]) };
    const unsigned sBb[2] = { lds_off(&sB[0][0]), lds_off(&sB[1][0]) };

#if USE_ASYNC
    // prologue: fill buffer 0 with K-step 0
    async_b128(sAb[0] + stA0, aP0);
    async_b128(sAb[0] + stA1, aP1);
    async_b128(sBb[0] + stA0, bP0);
    async_b128(sBb[0] + stA1, bP1);
    aP0 += BK; aP1 += BK; bP0 += BK; bP1 += BK;
    wait_async0();
    __syncthreads();
#else
    {
        *(uint4*)(&sA[0][rA0 * LDT + cA0 * 8]) = *(const uint4*)aP0;
        *(uint4*)(&sA[0][rA1 * LDT + cA1 * 8]) = *(const uint4*)aP1;
        *(uint4*)(&sB[0][rA0 * LDT + cA0 * 8]) = *(const uint4*)bP0;
        *(uint4*)(&sB[0][rA1 * LDT + cA1 * 8]) = *(const uint4*)bP1;
        aP0 += BK; aP1 += BK; bP0 += BK; bP1 += BK;
        __syncthreads();
    }
#endif

    // fragment element offsets within a buffer (constant per lane)
    const int fA0 = (waveM * 32 + mA) * LDT + kA;
    const int fA1 = (waveM * 32 + 16 + mA) * LDT + kA;
    int fB[4];
    #pragma unroll
    for (int nt = 0; nt < 4; ++nt)
        fB[nt] = (waveN * 64 + nt * 16 + nB) * LDT + kB;

    int buf = 0;
    for (int k0 = 0; k0 < K; k0 += BK) {
        const bool more = (k0 + BK) < K;
#if USE_ASYNC
        if (more) {
            const int nb = buf ^ 1;
            async_b128(sAb[nb] + stA0, aP0);
            async_b128(sAb[nb] + stA1, aP1);
            async_b128(sBb[nb] + stA0, bP0);
            async_b128(sBb[nb] + stA1, bP1);
            aP0 += BK; aP1 += BK; bP0 += BK; bP1 += BK;
        }
#else
        uint4 ra0, ra1, rb0, rb1;
        if (more) {
            ra0 = *(const uint4*)aP0; ra1 = *(const uint4*)aP1;
            rb0 = *(const uint4*)bP0; rb1 = *(const uint4*)bP1;
            aP0 += BK; aP1 += BK; bP0 += BK; bP1 += BK;
        }
#endif
        // ---- issue ALL fragment loads first, then compute ----
        union Frag { uint4 q[2]; v16bf v; };
        Frag a0, a1, b[4];
        const u16* pa = &sA[buf][0];
        const u16* pb = &sB[buf][0];
        a0.q[0] = *(const uint4*)(pa + fA0);
        a0.q[1] = *(const uint4*)(pa + fA0 + 16);
        a1.q[0] = *(const uint4*)(pa + fA1);
        a1.q[1] = *(const uint4*)(pa + fA1 + 16);
        #pragma unroll
        for (int nt = 0; nt < 4; ++nt) {
            b[nt].q[0] = *(const uint4*)(pb + fB[nt]);
            b[nt].q[1] = *(const uint4*)(pb + fB[nt] + 16);
        }
        #pragma unroll
        for (int nt = 0; nt < 4; ++nt) {
            acc[0][nt] = __builtin_amdgcn_wmma_f32_16x16x32_bf16(
                false, a0.v, false, b[nt].v, (short)0, acc[0][nt], false, false);
            acc[1][nt] = __builtin_amdgcn_wmma_f32_16x16x32_bf16(
                false, a1.v, false, b[nt].v, (short)0, acc[1][nt], false, false);
        }
        if (more) {
#if USE_ASYNC
            wait_async0();
#else
            __syncthreads();   // all reads of lds[buf^1] done before overwrite
            *(uint4*)(&sA[buf ^ 1][rA0 * LDT + cA0 * 8]) = ra0;
            *(uint4*)(&sA[buf ^ 1][rA1 * LDT + cA1 * 8]) = ra1;
            *(uint4*)(&sB[buf ^ 1][rA0 * LDT + cA0 * 8]) = rb0;
            *(uint4*)(&sB[buf ^ 1][rA1 * LDT + cA1 * 8]) = rb1;
#endif
            __syncthreads();
            buf ^= 1;
        }
    }

    // ---- epilogue (C/D layout: lane<16 -> M=r, lane>=16 -> M=8+r) ----
    #pragma unroll
    for (int mt = 0; mt < 2; ++mt) {
        const int rowBase = m0 + waveM * 32 + mt * 16 + hi * 8;
        #pragma unroll
        for (int nt = 0; nt < 4; ++nt) {
            const int col = n0 + waveN * 64 + nt * 16 + nB;
            const float b = bias ? bias[col] : 0.0f;
            #pragma unroll
            for (int r = 0; r < 8; ++r) {
                const int row = rowBase + r;
                if (row < M) {
                    float v = acc[mt][nt][r] + b;
                    size_t idx = (size_t)row * N + col;
                    if (MODE == 0) {
                        ((u16*)Cout)[idx] = f32_to_bf16(swishf(v));
                    } else if (MODE == 1) {
                        ((float*)Cout)[idx] = swishf(v);
                    } else if (MODE == 2) {
                        ((float*)Cout)[idx] = v;
                    } else {
                        ((float*)Cout)[idx] += swishf(v);
                    }
                }
            }
        }
    }
}

// ------------------------------------------------------------------
// Support kernels
// ------------------------------------------------------------------
__global__ void xpose_bf16(const float* __restrict__ W, u16* __restrict__ Wt,
                           int K, int N) {
    int idx = blockIdx.x * blockDim.x + threadIdx.x;
    if (idx >= K * N) return;
    int n = idx / K, k = idx - n * K;
    Wt[idx] = f32_to_bf16(W[(size_t)k * N + n]);   // Wt[n*K + k]
}

__global__ void zero_f32(float* __restrict__ p, int n) {
    int i = blockIdx.x * blockDim.x + threadIdx.x;
    if (i < n) p[i] = 0.0f;
}

__global__ void copy_f32(const float* __restrict__ a, float* __restrict__ b, int n) {
    int i = blockIdx.x * blockDim.x + threadIdx.x;
    if (i < n) b[i] = a[i];
}

__global__ void conv_bf16(const float* __restrict__ a, u16* __restrict__ b, int n) {
    int i = blockIdx.x * blockDim.x + threadIdx.x;
    if (i < n) b[i] = f32_to_bf16(a[i]);
}

__global__ void count_edges(const int* __restrict__ tgt, float* __restrict__ cnt, int E) {
    int e = blockIdx.x * blockDim.x + threadIdx.x;
    if (e < E) atomicAdd(&cnt[tgt[e]], 1.0f);
}

// feat[e] = concat(h[tgt[e]][0:384], h[src[e]][0:128]) -> bf16, width 512
__global__ void gather_feat(const float* __restrict__ h, const int* __restrict__ src,
                            const int* __restrict__ tgt, u16* __restrict__ feat, int E) {
    int idx = blockIdx.x * blockDim.x + threadIdx.x;
    if (idx >= E * 512) return;
    int e = idx >> 9, c = idx & 511;
    float v = (c < F_) ? h[(size_t)tgt[e] * F_ + c]
                       : h[(size_t)src[e] * F_ + (c - F_)];
    feat[idx] = f32_to_bf16(v);
}

// column sums / sum-of-squares over M rows (C == blockDim.x == 512)
__global__ __launch_bounds__(512)
void bn_stats(const float* __restrict__ x, float* __restrict__ stats, int M, int C) {
    int col = threadIdx.x;
    int r0 = blockIdx.x * 128;
    int r1 = r0 + 128; if (r1 > M) r1 = M;
    float s = 0.f, s2 = 0.f;
    for (int r = r0; r < r1; ++r) {
        float v = x[(size_t)r * C + col];
        s += v; s2 += v * v;
    }
    atomicAdd(&stats[col], s);
    atomicAdd(&stats[C + col], s2);
}

__global__ void bn_apply(const float* __restrict__ x, const float* __restrict__ stats,
                         const float* __restrict__ g, const float* __restrict__ be,
                         u16* __restrict__ out, int M, int C) {
    int idx = blockIdx.x * blockDim.x + threadIdx.x;
    if (idx >= M * C) return;
    int col = idx % C;
    float inv = 1.0f / (float)M;
    float mean = stats[col] * inv;
    float var  = stats[C + col] * inv - mean * mean;
    float v = x[idx];
    out[idx] = f32_to_bf16(g[col] * (v - mean) * rsqrtf(var + 1e-5f) + be[col]);
}

// agg[tgt[e]][c] += m[e][c]   (m: E x 256 f32)
__global__ void seg_sum(const float* __restrict__ m, const int* __restrict__ tgt,
                        float* __restrict__ agg, int E) {
    int idx = blockIdx.x * blockDim.x + threadIdx.x;
    if (idx >= E * 256) return;
    int e = idx >> 8, c = idx & 255;
    atomicAdd(&agg[(size_t)tgt[e] * 256 + c], m[idx]);
}

// U[n] = concat(x[n][0:128], h[n][0:384], agg[n]/max(cnt[n],1)) -> bf16, width 768
__global__ void build_u(const float* __restrict__ x, const float* __restrict__ h,
                        const float* __restrict__ agg, const float* __restrict__ cnt,
                        u16* __restrict__ U, int N) {
    int idx = blockIdx.x * blockDim.x + threadIdx.x;
    if (idx >= N * 768) return;
    int n = idx / 768, c = idx - n * 768;
    float v;
    if (c < V_)            v = x[(size_t)n * F_ + c];
    else if (c < V_ + F_)  v = h[(size_t)n * F_ + (c - V_)];
    else                   v = agg[(size_t)n * 256 + (c - V_ - F_)] / fmaxf(cnt[n], 1.0f);
    U[idx] = f32_to_bf16(v);
}

// column sums over M rows (C == blockDim.x == 256)
__global__ void colsum(const float* __restrict__ x, float* __restrict__ out, int M, int C) {
    int col = threadIdx.x;
    int r0 = blockIdx.x * 128;
    int r1 = r0 + 128; if (r1 > M) r1 = M;
    float s = 0.f;
    for (int r = r0; r < r1; ++r) s += x[(size_t)r * C + col];
    atomicAdd(&out[col], s);
}

// pooled(256 sums) -> /N -> swish(@qW1+qb1) -> @qW2+qb2 -> einsum with x1[:128]
__global__ __launch_bounds__(256)
void head_k(const float* __restrict__ pooled, int N,
            const float* __restrict__ qW1, const float* __restrict__ qb1,
            const float* __restrict__ qW2, const float* __restrict__ qb2,
            const float* __restrict__ x, float* __restrict__ out) {
    __shared__ float c1[256];
    __shared__ float c2[128];
    int t = threadIdx.x;
    float inv = 1.0f / (float)N;
    {
        float s = qb1[t];
        for (int k = 0; k < 256; ++k) s += (pooled[k] * inv) * qW1[k * 256 + t];
        c1[t] = swishf(s);
    }
    __syncthreads();
    if (t < 128) {
        float s = qb2[t];
        for (int k = 0; k < 256; ++k) s += c1[k] * qW2[k * 128 + t];
        c2[t] = s;
    }
    __syncthreads();
    if (t < 128) {
        float s = 0.f;
        for (int n = 0; n < 128; ++n) s += c2[n] * x[(size_t)n * F_ + t];
        out[t] = s;
    }
}

// ------------------------------------------------------------------
// Host orchestration
// ------------------------------------------------------------------
extern "C" void kernel_launch(void* const* d_in, const int* in_sizes, int n_in,
                              void* d_out, int out_size, void* d_ws, size_t ws_size,
                              hipStream_t stream) {
    (void)n_in; (void)out_size; (void)ws_size;

    const int N    = in_sizes[0] / F_;   // 10000
    const int E    = in_sizes[1] / 2;    // 100000
    const int MH   = 512, MO = 256;
    const int MIN_ = F_ + V_;            // 512
    const int UH   = 512;
    const int UIN  = MO + F_ + V_;       // 768
    const int PPH  = 512, PPO = 256;

    const float* x   = (const float*)d_in[0];
    const int*   src = (const int*)d_in[1];
    const int*   tgt = src + E;
    const float* mW1 = (const float*)d_in[2];
    const float* mb1 = (const float*)d_in[3];
    const float* g1  = (const float*)d_in[4];
    const float* be1 = (const float*)d_in[5];
    const float* mW2 = (const float*)d_in[6];
    const float* mb2 = (const float*)d_in[7];
    const float* g2  = (const float*)d_in[8];
    const float* be2 = (const float*)d_in[9];
    const float* mW3 = (const float*)d_in[10];
    const float* mb3 = (const float*)d_in[11];
    const float* g3  = (const float*)d_in[12];
    const float* be3 = (const float*)d_in[13];
    const float* mW4 = (const float*)d_in[14];
    const float* mb4 = (const float*)d_in[15];
    const float* uW1 = (const float*)d_in[16];
    const float* ub1 = (const float*)d_in[17];
    const float* uW2 = (const float*)d_in[18];
    const float* ub2 = (const float*)d_in[19];
    const float* uW3 = (const float*)d_in[20];
    const float* ub3 = (const float*)d_in[21];
    const float* uW4 = (const float*)d_in[22];
    const float* ub4 = (const float*)d_in[23];
    const float* pW1 = (const float*)d_in[24];
    const float* pb1 = (const float*)d_in[25];
    const float* pW2 = (const float*)d_in[26];
    const float* pb2 = (const float*)d_in[27];
    const float* pW3 = (const float*)d_in[28];
    const float* pb3 = (const float*)d_in[29];
    const float* pW4 = (const float*)d_in[30];
    const float* pb4 = (const float*)d_in[31];
    const float* qW1 = (const float*)d_in[32];
    const float* qb1 = (const float*)d_in[33];
    const float* qW2 = (const float*)d_in[34];
    const float* qb2 = (const float*)d_in[35];

    // bump allocator over workspace
    char* base = (char*)d_ws;
    size_t off = 0;
    auto alloc = [&](size_t bytes) -> void* {
        void* p = base + off;
        off += (bytes + 255) & ~(size_t)255;
        return p;
    };

    // weight convert+transpose (f32 KxN -> bf16 NxK)
    auto xpose = [&](const float* W, int K, int Nc) -> u16* {
        u16* Wt = (u16*)alloc((size_t)K * Nc * sizeof(u16));
        int total = K * Nc;
        xpose_bf16<<<(unsigned)((total + 255) / 256), 256, 0, stream>>>(W, Wt, K, Nc);
        return Wt;
    };

    u16 *mW1t[2], *mW2t[2], *mW3t[2], *mW4t[2];
    u16 *uW1t[2], *uW2t[2], *uW3t[2], *uW4t[2];
    for (int l = 0; l < 2; ++l) {
        mW1t[l] = xpose(mW1 + (size_t)l * MIN_ * MH, MIN_, MH);
        mW2t[l] = xpose(mW2 + (size_t)l * MH * MH,   MH,   MH);
        mW3t[l] = xpose(mW3 + (size_t)l * MH * MH,   MH,   MH);
        mW4t[l] = xpose(mW4 + (size_t)l * MH * MO,   MH,   MO);
        uW1t[l] = xpose(uW1 + (size_t)l * UIN * UH,  UIN,  UH);
        uW2t[l] = xpose(uW2 + (size_t)l * UH * UH,   UH,   UH);
        uW3t[l] = xpose(uW3 + (size_t)l * UH * UH,   UH,   UH);
        uW4t[l] = xpose(uW4 + (size_t)l * UH * F_,   UH,   F_);
    }
    u16* pW1t = xpose(pW1, F_,  PPH);
    u16* pW2t = xpose(pW2, PPH, PPH);
    u16* pW3t = xpose(pW3, PPH, PPH);
    u16* pW4t = xpose(pW4, PPH, PPO);

    u16*   featb  = (u16*)  alloc((size_t)E * 512 * sizeof(u16));  // edge activations (bf16)
    float* t1     = (float*)alloc((size_t)E * 512 * sizeof(float));// edge f32 scratch
    float* h      = (float*)alloc((size_t)N * F_ * sizeof(float));
    float* agg    = (float*)alloc((size_t)N * MO * sizeof(float)); // also reused as p4 f32
    float* cnt    = (float*)alloc((size_t)N * sizeof(float));
    u16*   Ub     = (u16*)  alloc((size_t)N * UIN * sizeof(u16));  // also reused as hbf16
    u16*   bufA   = (u16*)  alloc((size_t)N * 512 * sizeof(u16));
    u16*   bufB   = (u16*)  alloc((size_t)N * 512 * sizeof(u16));
    float* stats  = (float*)alloc(1024 * sizeof(float));
    float* pooled = (float*)alloc(256 * sizeof(float));

    auto zero = [&](float* p, int n) {
        zero_f32<<<(unsigned)((n + 255) / 256), 256, 0, stream>>>(p, n);
    };
    auto gemm = [&](int mode, const u16* A, const u16* Bt, const float* bias,
                    void* C, int M, int Nc, int K) {
        dim3 grid((unsigned)((M + BM - 1) / BM), (unsigned)(Nc / BNC));
        switch (mode) {
        case 0:  wmma_gemm<0><<<grid, 256, 0, stream>>>(A, Bt, bias, C, M, Nc, K); break;
        case 1:  wmma_gemm<1><<<grid, 256, 0, stream>>>(A, Bt, bias, C, M, Nc, K); break;
        case 2:  wmma_gemm<2><<<grid, 256, 0, stream>>>(A, Bt, bias, C, M, Nc, K); break;
        default: wmma_gemm<3><<<grid, 256, 0, stream>>>(A, Bt, bias, C, M, Nc, K); break;
        }
    };

    // degree counts (tgt fixed across layers)
    zero(cnt, N);
    count_edges<<<(unsigned)((E + 255) / 256), 256, 0, stream>>>(tgt, cnt, E);
    // h = x
    copy_f32<<<(unsigned)((N * F_ + 255) / 256), 256, 0, stream>>>(x, h, N * F_);

    const unsigned gbE512 = (unsigned)(((size_t)E * 512 + 255) / 256);
    const unsigned gbE256 = (unsigned)(((size_t)E * 256 + 255) / 256);
    const unsigned gbStats = (unsigned)((E + 127) / 128);

    for (int l = 0; l < 2; ++l) {
        // --- message MLP over edges ---
        gather_feat<<<gbE512, 256, 0, stream>>>(h, src, tgt, featb, E);

        gemm(1, featb, mW1t[l], mb1 + l * MH, t1, E, MH, MIN_);
        zero(stats, 2 * MH);
        bn_stats<<<gbStats, 512, 0, stream>>>(t1, stats, E, MH);
        bn_apply<<<gbE512, 256, 0, stream>>>(t1, stats, g1 + l * MH, be1 + l * MH, featb, E, MH);

        gemm(1, featb, mW2t[l], mb2 + l * MH, t1, E, MH, MH);
        zero(stats, 2 * MH);
        bn_stats<<<gbStats, 512, 0, stream>>>(t1, stats, E, MH);
        bn_apply<<<gbE512, 256, 0, stream>>>(t1, stats, g2 + l * MH, be2 + l * MH, featb, E, MH);

        gemm(1, featb, mW3t[l], mb3 + l * MH, t1, E, MH, MH);
        zero(stats, 2 * MH);
        bn_stats<<<gbStats, 512, 0, stream>>>(t1, stats, E, MH);
        bn_apply<<<gbE512, 256, 0, stream>>>(t1, stats, g3 + l * MH, be3 + l * MH, featb, E, MH);

        gemm(1, featb, mW4t[l], mb4 + l * MO, t1, E, MO, MH);   // t1: E x 256 f32 (swish)

        // --- mean aggregation ---
        zero(agg, N * MO);
        seg_sum<<<gbE256, 256, 0, stream>>>(t1, tgt, agg, E);

        // --- update MLP over nodes, residual into h ---
        build_u<<<(unsigned)(((size_t)N * UIN + 255) / 256), 256, 0, stream>>>(x, h, agg, cnt, Ub, N);
        gemm(0, Ub,   uW1t[l], ub1 + l * UH, bufA, N, UH, UIN);
        gemm(0, bufA, uW2t[l], ub2 + l * UH, bufB, N, UH, UH);
        gemm(0, bufB, uW3t[l], ub3 + l * UH, bufA, N, UH, UH);
        gemm(3, bufA, uW4t[l], ub4 + l * F_, h,    N, F_, UH);
    }

    // --- post MLP + pooling + head ---
    conv_bf16<<<(unsigned)((N * F_ + 255) / 256), 256, 0, stream>>>(h, Ub, N * F_);
    gemm(0, Ub,   pW1t, pb1, bufA, N, PPH, F_);
    gemm(0, bufA, pW2t, pb2, bufB, N, PPH, PPH);
    gemm(0, bufB, pW3t, pb3, bufA, N, PPH, PPH);
    gemm(2, bufA, pW4t, pb4, agg,  N, PPO, PPH);   // agg reused: N x 256 f32

    zero(pooled, 256);
    colsum<<<(unsigned)((N + 127) / 128), 256, 0, stream>>>(agg, pooled, N, PPO);
    head_k<<<1, 256, 0, stream>>>(pooled, N, qW1, qb1, qW2, qb2, x, (float*)d_out);
}
